// Encoder_37684043055391
// MI455X (gfx1250) — matrix operand order repaired and checked by reading, not compiled, not run on hardware
//
#include <hip/hip_runtime.h>
#include <hip/hip_bf16.h>
#include <math.h>

// ---------------------------------------------------------------------------
// 2-layer bidirectional GRU (T=4096, H=512, EMB=100) for gfx1250 / MI455X.
//  * Time-batched input projections  -> v_wmma_f32_16x16x32_f16 GEMMs
//  * Sequential recurrence           -> WMMA matvec (M=1 padded to 16),
//                                       f16 W_hh streamed from L2 (native
//                                       N x K layout), h in LDS, f32 gates
// ---------------------------------------------------------------------------

#define T_LEN   4096
#define EMB_IN  100
#define EMB_PAD 128
#define HID     512
#define G3      (3 * HID)          // 1536
#define OUT_W   (2 * HID)          // 1024

typedef __attribute__((ext_vector_type(16))) _Float16 v16h;
typedef __attribute__((ext_vector_type(8)))  _Float16 v8h;
typedef __attribute__((ext_vector_type(8)))  float    v8f;

// ------------------------- prep kernels ------------------------------------

// embedding gather -> f16, K padded 100 -> 128 with zeros
__global__ void gru_embed_pad_f16(const int* __restrict__ words,
                                  const float* __restrict__ emb,
                                  _Float16* __restrict__ xh) {
    const int t = blockIdx.x;            // 0..T-1
    const int e = threadIdx.x;           // 0..127
    float v = 0.0f;
    if (e < EMB_IN) v = emb[(size_t)words[t] * EMB_IN + e];
    xh[(size_t)t * EMB_PAD + e] = (_Float16)v;
}

// f32 (rows x kin) row-major -> f16 (rows x kpad) row-major, zero pad
__global__ void gru_conv_pad_f16(const float* __restrict__ src,
                                 _Float16* __restrict__ dst,
                                 int rows, int kin, int kpad) {
    const int idx = blockIdx.x * blockDim.x + threadIdx.x;
    if (idx >= rows * kpad) return;
    const int r = idx / kpad, k = idx - r * kpad;
    float v = (k < kin) ? src[(size_t)r * kin + k] : 0.0f;
    dst[idx] = (_Float16)v;
}

__global__ void gru_f32_to_f16(const float* __restrict__ s,
                               _Float16* __restrict__ d, int n) {
    const int idx = blockIdx.x * blockDim.x + threadIdx.x;
    if (idx < n) d[idx] = (_Float16)s[idx];
}

// ------------------------- WMMA GEMM ---------------------------------------
// C[M,N] (f32) = A[M,K] (f16 row-major) x W[N,K]^T (f16 row-major) + bias[N]
// One 16x16 output tile per wave; K stepped by 32 through v_wmma_f32_16x16x32_f16.
__global__ __launch_bounds__(128)
void gru_wmma_gemm_bias(const _Float16* __restrict__ A,
                        const _Float16* __restrict__ W,
                        const float* __restrict__ bias,
                        float* __restrict__ C,
                        int M, int N, int K) {
    const int lane = threadIdx.x & 31;
    const int wave = threadIdx.x >> 5;
    const int tiles_n = N >> 4;
    const int tile = blockIdx.x * (blockDim.x >> 5) + wave;
    const int tm = tile / tiles_n;
    const int tn = tile - tm * tiles_n;
    if (tm >= (M >> 4)) return;                 // wave-uniform guard (EXEC stays full)

    const int q  = lane & 15;                   // A: row m ; B: col n ; D: col n
    const int hi = lane >> 4;

    const _Float16* arow = A + (size_t)(tm * 16 + q) * K;   // A fragment source
    const _Float16* wrow = W + (size_t)(tn * 16 + q) * K;   // B fragment source

    v8f acc = {};
    #pragma unroll 4
    for (int kk = 0; kk < K; kk += 32) {
        // A lane layout: halves 0..7 -> K = kk+8*hi .. +7 ; 8..15 -> K = kk+16+8*hi .. +7
        v8h a0 = *(const v8h*)(arow + kk + 8 * hi);
        v8h a1 = *(const v8h*)(arow + kk + 16 + 8 * hi);
        // B lane layout: same as A (0..7 -> K = kk+8*hi .. +7 ; 8..15 -> K = kk+16+8*hi .. +7)
        v8h b0 = *(const v8h*)(wrow + kk + 8 * hi);
        v8h b1 = *(const v8h*)(wrow + kk + 16 + 8 * hi);
        v16h av, bv;
        #pragma unroll
        for (int i = 0; i < 8; ++i) {
            av[i] = a0[i]; av[i + 8] = a1[i];
            bv[i] = b0[i]; bv[i + 8] = b1[i];
        }
        acc = __builtin_amdgcn_wmma_f32_16x16x32_f16(
            false, av, false, bv, (short)0, acc, false, false);
    }

    // D layout: VGPR v -> row (v + 8*hi), col q
    const float bn = bias[tn * 16 + q];
    float* crow = C + (size_t)(tm * 16 + 8 * hi) * N + tn * 16 + q;
    #pragma unroll
    for (int v = 0; v < 8; ++v)
        crow[(size_t)v * N] = acc[v] + bn;
}

// ------------------------- recurrence (WMMA matvec) -------------------------
// One 512-thread workgroup (16 waves) per direction (gridDim.x == 2).
// Per step: gh[0:1536] = h(1x512,f16) x W_hh^T via WMMA tiles with only row
// M=0 populated; wave w owns column tiles w*6 .. w*6+5. W_hh is streamed in
// its native (3H x 512) row-major layout as f16 (L2-resident, 1.5 MB/step).
// Then 512 threads apply the GRU gate math in f32.
__global__ __launch_bounds__(512)
void gru_recurrence_wmma(const float* __restrict__ gi_f, const float* __restrict__ gi_b,
                         const _Float16* __restrict__ wh_f, const _Float16* __restrict__ wh_b,
                         const float* __restrict__ bhh_f, const float* __restrict__ bhh_b,
                         float* __restrict__ y, int ystride,
                         float* __restrict__ state_f, float* __restrict__ state_b) {
    const int dir = blockIdx.x;                   // 0 = fwd, 1 = bwd
    const float*    gi  = dir ? gi_b  : gi_f;
    const _Float16* wh  = dir ? wh_b  : wh_f;
    const float*    bhh = dir ? bhh_b : bhh_f;
    float* st           = dir ? state_b : state_f;

    const int tid  = threadIdx.x;                 // 0..511 (owns hidden unit tid)
    const int lane = tid & 31;
    const int wave = tid >> 5;                    // 0..15
    const int q    = lane & 15;
    const int hi   = lane >> 4;

    __shared__ __align__(16) float    hs[HID];    // h (f32, carried state)
    __shared__ __align__(16) _Float16 h16[HID];   // h (f16, WMMA A source)
    __shared__ __align__(16) float    gh[G3];     // h @ W_hh^T for this step
    __shared__ __align__(16) _Float16 zpad[32];   // zero stub for A rows 1..15

    hs[tid] = 0.0f;
    h16[tid] = (_Float16)0.0f;
    if (tid < 32) zpad[tid] = (_Float16)0.0f;
    __syncthreads();

    const float br = bhh[tid], bz = bhh[HID + tid], bn = bhh[2 * HID + tid];

    for (int s = 0; s < T_LEN; ++s) {
        const int t = dir ? (T_LEN - 1 - s) : s;

        // ---- gh = h x W_hh^T via WMMA (row 0 of each 16x16 tile) ----
        #pragma unroll
        for (int tt = 0; tt < 6; ++tt) {
            const int tn = wave * 6 + tt;         // column tile (wave-uniform)
            const _Float16* wrow = wh + (size_t)(tn * 16 + q) * HID;
            v8f acc = {};
            #pragma unroll 4
            for (int kk = 0; kk < HID; kk += 32) {
                // B fragment: K runs kk+8*hi and kk+16+8*hi from W_hh row (tn*16+q), matching A
                v8h b0 = *(const v8h*)(wrow + kk + 8 * hi);
                v8h b1 = *(const v8h*)(wrow + kk + 16 + 8 * hi);
                // A fragment: row M=0 carries h; rows 1..15 read the zero stub
                const _Float16* pa0 = (q == 0) ? (h16 + kk + 8 * hi)      : zpad;
                const _Float16* pa1 = (q == 0) ? (h16 + kk + 16 + 8 * hi) : zpad;
                v8h a0 = *(const v8h*)pa0;
                v8h a1 = *(const v8h*)pa1;
                v16h av, bv;
                #pragma unroll
                for (int i = 0; i < 8; ++i) {
                    av[i] = a0[i]; av[i + 8] = a1[i];
                    bv[i] = b0[i]; bv[i + 8] = b1[i];
                }
                acc = __builtin_amdgcn_wmma_f32_16x16x32_f16(
                    false, av, false, bv, (short)0, acc, false, false);
            }
            // D row M=0 lives in VGPR0 of lanes 0..15 (N = lane)
            if (lane < 16) gh[tn * 16 + lane] = acc[0];
        }
        __syncthreads();                          // gh complete; matvec reads of h16 done

        // ---- elementwise GRU gate update (f32) ----
        const float* g = gi + (size_t)t * G3;
        const float r  = 1.0f / (1.0f + __expf(-(g[tid]           + gh[tid]           + br)));
        const float z  = 1.0f / (1.0f + __expf(-(g[HID + tid]     + gh[HID + tid]     + bz)));
        const float nn = tanhf(g[2 * HID + tid] + r * (gh[2 * HID + tid] + bn));
        const float hn = (1.0f - z) * nn + z * hs[tid];

        y[(size_t)t * ystride + dir * HID + tid] = hn;

        hs[tid]  = hn;
        h16[tid] = (_Float16)hn;
        __syncthreads();                          // new h visible before next matvec
    }
    st[tid] = hs[tid];
}

// ------------------------- launcher ----------------------------------------

extern "C" void kernel_launch(void* const* d_in, const int* in_sizes, int n_in,
                              void* d_out, int out_size, void* d_ws, size_t ws_size,
                              hipStream_t stream) {
    const int*   words = (const int*)  d_in[0];
    const float* emb   = (const float*)d_in[1];
    // per (layer,dir): w_ih, w_hh, b_ih, b_hh
    const float* w_ih_l0f = (const float*)d_in[2];
    const float* w_hh_l0f = (const float*)d_in[3];
    const float* b_ih_l0f = (const float*)d_in[4];
    const float* b_hh_l0f = (const float*)d_in[5];
    const float* w_ih_l0b = (const float*)d_in[6];
    const float* w_hh_l0b = (const float*)d_in[7];
    const float* b_ih_l0b = (const float*)d_in[8];
    const float* b_hh_l0b = (const float*)d_in[9];
    const float* w_ih_l1f = (const float*)d_in[10];
    const float* w_hh_l1f = (const float*)d_in[11];
    const float* b_ih_l1f = (const float*)d_in[12];
    const float* b_hh_l1f = (const float*)d_in[13];
    const float* w_ih_l1b = (const float*)d_in[14];
    const float* w_hh_l1b = (const float*)d_in[15];
    const float* b_ih_l1b = (const float*)d_in[16];
    const float* b_hh_l1b = (const float*)d_in[17];

    float* out = (float*)d_out;

    // ---- carve workspace (256B aligned slices) ----
    char* p = (char*)d_ws;
    auto carve = [&](size_t bytes) -> char* {
        char* r = p;
        p += (bytes + 255) & ~(size_t)255;
        return r;
    };
    _Float16* x_h16   = (_Float16*)carve((size_t)T_LEN * EMB_PAD * 2);
    _Float16* wih0f_h = (_Float16*)carve((size_t)G3 * EMB_PAD * 2);
    _Float16* wih0b_h = (_Float16*)carve((size_t)G3 * EMB_PAD * 2);
    _Float16* wih1f_h = (_Float16*)carve((size_t)G3 * OUT_W * 2);
    _Float16* wih1b_h = (_Float16*)carve((size_t)G3 * OUT_W * 2);
    _Float16* wh0f_h  = (_Float16*)carve((size_t)G3 * HID * 2);
    _Float16* wh0b_h  = (_Float16*)carve((size_t)G3 * HID * 2);
    _Float16* wh1f_h  = (_Float16*)carve((size_t)G3 * HID * 2);
    _Float16* wh1b_h  = (_Float16*)carve((size_t)G3 * HID * 2);
    float*    gi_f    = (float*)   carve((size_t)T_LEN * G3 * 4);
    float*    gi_b    = (float*)   carve((size_t)T_LEN * G3 * 4);
    float*    y0      = (float*)   carve((size_t)T_LEN * OUT_W * 4);
    _Float16* y0_h16  = (_Float16*)carve((size_t)T_LEN * OUT_W * 2);

    // ---- one-time format prep ----
    gru_embed_pad_f16<<<T_LEN, EMB_PAD, 0, stream>>>(words, emb, x_h16);

    {
        const int n0 = G3 * EMB_PAD;                  // 196608
        gru_conv_pad_f16<<<(n0 + 255) / 256, 256, 0, stream>>>(w_ih_l0f, wih0f_h, G3, EMB_IN, EMB_PAD);
        gru_conv_pad_f16<<<(n0 + 255) / 256, 256, 0, stream>>>(w_ih_l0b, wih0b_h, G3, EMB_IN, EMB_PAD);
        const int n1 = G3 * OUT_W;                    // 1572864
        gru_conv_pad_f16<<<(n1 + 255) / 256, 256, 0, stream>>>(w_ih_l1f, wih1f_h, G3, OUT_W, OUT_W);
        gru_conv_pad_f16<<<(n1 + 255) / 256, 256, 0, stream>>>(w_ih_l1b, wih1b_h, G3, OUT_W, OUT_W);
        const int nh = G3 * HID;                      // 786432 (no transpose: native N x K)
        gru_conv_pad_f16<<<(nh + 255) / 256, 256, 0, stream>>>(w_hh_l0f, wh0f_h, G3, HID, HID);
        gru_conv_pad_f16<<<(nh + 255) / 256, 256, 0, stream>>>(w_hh_l0b, wh0b_h, G3, HID, HID);
        gru_conv_pad_f16<<<(nh + 255) / 256, 256, 0, stream>>>(w_hh_l1f, wh1f_h, G3, HID, HID);
        gru_conv_pad_f16<<<(nh + 255) / 256, 256, 0, stream>>>(w_hh_l1b, wh1b_h, G3, HID, HID);
    }

    // ---- layer 0: time-batched input GEMMs (WMMA), then recurrence ----
    const int tilesM = T_LEN / 16, tilesN = G3 / 16;          // 256 x 96
    const int gemm_blocks = (tilesM * tilesN) / 4;            // 4 waves / block
    gru_wmma_gemm_bias<<<gemm_blocks, 128, 0, stream>>>(x_h16, wih0f_h, b_ih_l0f, gi_f,
                                                        T_LEN, G3, EMB_PAD);
    gru_wmma_gemm_bias<<<gemm_blocks, 128, 0, stream>>>(x_h16, wih0b_h, b_ih_l0b, gi_b,
                                                        T_LEN, G3, EMB_PAD);

    const size_t S = (size_t)T_LEN * OUT_W;   // state offset in d_out
    gru_recurrence_wmma<<<2, HID, 0, stream>>>(gi_f, gi_b, wh0f_h, wh0b_h, b_hh_l0f, b_hh_l0b,
                                               y0, OUT_W,
                                               out + S + 0 * HID,   // hf0
                                               out + S + 1 * HID);  // hb0

    // ---- layer 1 ----
    {
        const int n = T_LEN * OUT_W;
        gru_f32_to_f16<<<(n + 255) / 256, 256, 0, stream>>>(y0, y0_h16, n);
    }
    gru_wmma_gemm_bias<<<gemm_blocks, 128, 0, stream>>>(y0_h16, wih1f_h, b_ih_l1f, gi_f,
                                                        T_LEN, G3, OUT_W);
    gru_wmma_gemm_bias<<<gemm_blocks, 128, 0, stream>>>(y0_h16, wih1b_h, b_ih_l1b, gi_b,
                                                        T_LEN, G3, OUT_W);

    gru_recurrence_wmma<<<2, HID, 0, stream>>>(gi_f, gi_b, wh1f_h, wh1b_h, b_hh_l1f, b_hh_l1b,
                                               out, OUT_W,
                                               out + S + 2 * HID,   // hf1
                                               out + S + 3 * HID);  // hb1
}
